// attention_9526237463064
// MI455X (gfx1250) — matrix operand order repaired and checked by reading, not compiled, not run on hardware
//
#include <hip/hip_runtime.h>
#include <hip/hip_bf16.h>

typedef _Float16 f16;
typedef __attribute__((ext_vector_type(16))) _Float16 v16h;
typedef __attribute__((ext_vector_type(8)))  _Float16 v8h;
typedef __attribute__((ext_vector_type(8)))  float    v8f;
typedef __attribute__((ext_vector_type(4)))  unsigned int v4u;
typedef __attribute__((ext_vector_type(4)))  int v4i;
typedef __attribute__((ext_vector_type(8)))  int v8i;

#define Bsz   4096
#define Tlen  200
#define Dd    64
#define CH    48       // chunk of T-rows per iteration (3 M-tiles)
#define NCH   5        // 5*48 = 240 >= 200
#define TPAD  (CH*NCH)

// workspace layout: [W1^T 80x256 f16][W2^T 48x96 f16][f32: b1(80) b2(40) W3(40) b3(1)]
#define WS_W1T_HALFS (80 * 256)
#define WS_W2T_HALFS (48 * 96)
#define WS_F32_OFF   ((WS_W1T_HALFS + WS_W2T_HALFS) * 2)   // bytes, 50176 (16B aligned)

// ---------- fragment loaders (CDNA5 wave32 WMMA 16-bit layouts) ----------
// A 16x32 f16: lane L -> row M = L%16, half = L/16;
//   v16h elems 0..7  = K[k0 + half*8 .. +7]
//   v16h elems 8..15 = K[k0 + 16 + half*8 .. +7]
__device__ __forceinline__ v16h load_afrag(const f16* p /* = row + k0 + half*8 */) {
    union { v16h v; v8h h[2]; } u;
    u.h[0] = *(const v8h*)(p);
    u.h[1] = *(const v8h*)(p + 16);
    return u.v;
}
// B 32x16 f16: lane L -> col N = L%16, half = L/16; K[k0+half*8..+7], K[k0+16+half*8..+7]
__device__ __forceinline__ v16h load_bfrag(const f16* p /* = colrow + k0 + half*8 */) {
    union { v16h v; v8h h[2]; } u;
    u.h[0] = *(const v8h*)(p);
    u.h[1] = *(const v8h*)(p + 16);
    return u.v;
}

__device__ __forceinline__ float sigmoidf_fast(float x) {
    return 1.0f / (1.0f + __expf(-x));
}

// ---------- TDM: DMA a [rows<=CH] x 64 f32 tile of keys into LDS ----------
// D# per CDNA5 ISA ch.8: group0 = {count/lds_addr/global_addr/type},
// group1 = {data_size, tensor dims, tile dims, strides}. tensor_dim1 =
// remaining valid rows so out-of-bounds rows are zero-filled by the TDM.
__device__ __forceinline__ void tdm_load_keys_tile(const float* gsrc,
                                                   unsigned ldsOff, int rowsValid) {
    unsigned long long ga = (unsigned long long)(uintptr_t)gsrc;
    v4u g0;
    g0.x = 1u;                                                  // count=1, user desc
    g0.y = ldsOff;                                              // lds_addr (bytes)
    g0.z = (unsigned)(ga & 0xFFFFFFFFu);                        // global_addr[31:0]
    g0.w = (unsigned)((ga >> 32) & 0x1FFFFFFu) | (2u << 30);    // [56:32] | type=2
    v8i g1;
    g1[0] = (int)(2u << 16);                                    // data_size = 4B
    g1[1] = (int)(64u << 16);                                   // tensor_dim0 = 64
    g1[2] = (int)(((unsigned)rowsValid & 0xFFFFu) << 16);       // tensor_dim1 = rows
    g1[3] = (int)(64u << 16);                                   // tile_dim0 = 64
    g1[4] = (int)CH;                                            // tile_dim1=48, dim2=0
    g1[5] = 64;                                                 // tensor_dim0_stride
    g1[6] = 0;
    g1[7] = 0;
    v4i z4 = {0, 0, 0, 0};
#if __clang_major__ >= 23
    v8i z8 = {0, 0, 0, 0, 0, 0, 0, 0};
    __builtin_amdgcn_tensor_load_to_lds(g0, g1, z4, z4, z8, 0);
#else
    __builtin_amdgcn_tensor_load_to_lds(g0, g1, z4, z4, 0);
#endif
}

// ---------- prep: transpose + f16-convert weights into workspace ----------
__global__ void din_prep_weights(const float* __restrict__ W1, const float* __restrict__ b1,
                                 const float* __restrict__ W2, const float* __restrict__ b2,
                                 const float* __restrict__ W3, const float* __restrict__ b3,
                                 f16* __restrict__ wH, float* __restrict__ wF) {
    const int tid = threadIdx.x;
    f16* w1t = wH;                       // [80][256], w1t[n*256+k] = W1[k][n]
    f16* w2t = wH + WS_W1T_HALFS;        // [48][96],  zero padded
    for (int i = tid; i < 80 * 256; i += 256) {
        int n = i >> 8, k = i & 255;
        w1t[i] = (f16)W1[k * 80 + n];
    }
    for (int i = tid; i < 48 * 96; i += 256) {
        int n = i / 96, k = i - n * 96;
        w2t[i] = (n < 40 && k < 80) ? (f16)W2[k * 40 + n] : (f16)0.0f;
    }
    if (tid < 80) wF[tid] = b1[tid];
    if (tid < 40) { wF[80 + tid] = b2[tid]; wF[120 + tid] = W3[tid]; }
    if (tid == 0) wF[160] = b3[0];
}

// ---------- main: one block (8 wave32) per batch row ----------
__global__ __launch_bounds__(256)
void din_attention_kernel(const float* __restrict__ queries,
                          const float* __restrict__ keys,
                          const int*   __restrict__ keys_length,
                          const f16*   __restrict__ wH,
                          const float* __restrict__ wF,
                          float*       __restrict__ out) {
    __shared__ f16   sDin[CH][256];      // 24 KB : [q | k | q-k | q*k] f16
    __shared__ f16   sH1[CH][96];        //  9 KB : layer1 out, K-padded 80->96
    __shared__ float sH2[CH][40];        //7.5 KB : layer2 out
    __shared__ float sStage[CH * 64];    // 12 KB : TDM keys staging (f32)
    __shared__ float sLogits[TPAD];
    __shared__ float sQ[64];
    __shared__ float sB1[80];
    __shared__ float sB2[40];
    __shared__ float sW3[40];
    __shared__ float sMS[2];
    __shared__ float sRed[256];

    const int b     = blockIdx.x;
    const int tid   = threadIdx.x;
    const int lane  = tid & 31;          // wave32
    const int wave  = tid >> 5;          // 8 waves
    const int laneM = lane & 15;
    const int half  = lane >> 4;

    const f16* w1t = wH;
    const f16* w2t = wH + WS_W1T_HALFS;
    const float b3v = wF[160];

    const float* keysB = keys + (size_t)b * Tlen * Dd;
    // LDS byte offset of staging buffer (flat-LDS aperture keeps offset in addr[31:0])
    const unsigned stageOff = (unsigned)(uintptr_t)(&sStage[0]);

    // kick DMA for chunk 0 while we load q/biases
    if (wave == 0) tdm_load_keys_tile(keysB, stageOff, Tlen);

    if (tid < 64) sQ[tid] = queries[(size_t)b * Dd + tid];
    if (tid < 80) sB1[tid] = wF[tid];
    if (tid < 40) { sB2[tid] = wF[80 + tid]; sW3[tid] = wF[120 + tid]; }

    for (int chunk = 0; chunk < NCH; ++chunk) {
        const int t0 = chunk * CH;

        if (wave == 0) __builtin_amdgcn_s_wait_tensorcnt(0);
        __syncthreads();                 // staging tile (and q/biases) visible to all

        // build din chunk in f16 from staged keys
        for (int i = tid; i < CH * 64; i += 256) {
            int tl = i >> 6, d = i & 63;
            float kv = sStage[i];        // OOB rows already zero-filled by TDM
            float qv = sQ[d];
            sDin[tl][d]        = (f16)qv;
            sDin[tl][64 + d]   = (f16)kv;
            sDin[tl][128 + d]  = (f16)(qv - kv);
            sDin[tl][192 + d]  = (f16)(qv * kv);
        }
        __syncthreads();                 // staging consumed, din ready

        // overlap next chunk's keys DMA with the WMMA layers
        if (chunk + 1 < NCH && wave == 0)
            tdm_load_keys_tile(keysB + (size_t)(t0 + CH) * Dd, stageOff,
                               Tlen - (t0 + CH));

        // ---- layer 1: [48,256] x [256,80], 3x5 tiles over 8 waves ----
        for (int tile = wave; tile < 15; tile += 8) {
            int mt = tile / 5, nt = tile - mt * 5;
            v8f acc = {};
            int n = nt * 16 + laneM;
            const f16* arow = &sDin[mt * 16 + laneM][0];
            const f16* brow = w1t + (size_t)n * 256;
            #pragma unroll
            for (int kk = 0; kk < 8; ++kk) {
                int k0 = kk * 32;
                v16h a  = load_afrag(arow + k0 + half * 8);
                v16h bb = load_bfrag(brow + k0 + half * 8);
                acc = __builtin_amdgcn_wmma_f32_16x16x32_f16(
                          false, a, false, bb, (short)0, acc, false, false);
            }
            float bn = sB1[n];
            int mbase = mt * 16 + half * 8;     // C/D layout: M = r + 8*half
            #pragma unroll
            for (int r = 0; r < 8; ++r)
                sH1[mbase + r][n] = (f16)sigmoidf_fast(acc[r] + bn);
        }
        // zero the K padding (cols 80..95) for layer-2 WMMA
        for (int i = tid; i < CH * 16; i += 256)
            sH1[i >> 4][80 + (i & 15)] = (f16)0.0f;
        __syncthreads();

        // ---- layer 2: [48,96] x [96,48(pad)], 3x3 tiles over 8 waves ----
        for (int tile = wave; tile < 9; tile += 8) {
            int mt = tile / 3, nt = tile - mt * 3;
            v8f acc = {};
            int n = nt * 16 + laneM;
            const f16* arow = &sH1[mt * 16 + laneM][0];
            const f16* brow = w2t + (size_t)n * 96;
            #pragma unroll
            for (int kk = 0; kk < 3; ++kk) {
                int k0 = kk * 32;
                v16h a  = load_afrag(arow + k0 + half * 8);
                v16h bb = load_bfrag(brow + k0 + half * 8);
                acc = __builtin_amdgcn_wmma_f32_16x16x32_f16(
                          false, a, false, bb, (short)0, acc, false, false);
            }
            if (n < 40) {
                float bn = sB2[n];
                int mbase = mt * 16 + half * 8;
                #pragma unroll
                for (int r = 0; r < 8; ++r)
                    sH2[mbase + r][n] = sigmoidf_fast(acc[r] + bn);
            }
        }
        __syncthreads();

        // ---- layer 3: [48,40] x [40,1] scalar dot ----
        if (tid < CH) {
            float l = b3v;
            #pragma unroll 8
            for (int j = 0; j < 40; ++j) l += sH2[tid][j] * sW3[j];
            sLogits[t0 + tid] = l;
        }
        __syncthreads();
    }

    // warm L2/L0 for the final attn.keys pass (global_prefetch_b8)
    if (tid < Tlen) {
        __builtin_prefetch(&keysB[(size_t)tid * Dd], 0, 2);
        __builtin_prefetch(&keysB[(size_t)tid * Dd + 32], 0, 2);
    }

    // ---- masked, scaled softmax over T ----
    const int len = keys_length[b];
    if (tid < Tlen) {
        float l = sLogits[tid];
        l = (tid < len) ? l : -4294967295.0f;   // NEG_INF = -2^32+1
        sLogits[tid] = l * 0.125f;              // 1/sqrt(64)
    }
    __syncthreads();
    if (tid == 0) {
        float m = -3.4e38f;
        for (int t = 0; t < Tlen; ++t) m = fmaxf(m, sLogits[t]);
        sMS[0] = m;
    }
    __syncthreads();
    const float mx = sMS[0];
    if (tid < Tlen) sLogits[tid] = __expf(sLogits[tid] - mx);
    __syncthreads();
    if (tid == 0) {
        float s = 0.0f;
        for (int t = 0; t < Tlen; ++t) s += sLogits[t];
        sMS[1] = s;
    }
    __syncthreads();
    const float inv = 1.0f / sMS[1];

    // ---- out[b] = attn . keys  (keys re-read f32, L2-resident) ----
    const int g = tid >> 6, d = tid & 63;
    float p = 0.0f;
    for (int t = g; t < Tlen; t += 4)
        p += sLogits[t] * keysB[(size_t)t * Dd + d];
    sRed[tid] = p * inv;
    __syncthreads();
    if (tid < 64)
        out[(size_t)b * Dd + tid] =
            sRed[tid] + sRed[64 + tid] + sRed[128 + tid] + sRed[192 + tid];
}

extern "C" void kernel_launch(void* const* d_in, const int* in_sizes, int n_in,
                              void* d_out, int out_size, void* d_ws, size_t ws_size,
                              hipStream_t stream) {
    const float* queries     = (const float*)d_in[0];
    const float* keys        = (const float*)d_in[1];
    const int*   keys_length = (const int*)d_in[2];
    const float* W1          = (const float*)d_in[3];
    const float* b1          = (const float*)d_in[4];
    const float* W2          = (const float*)d_in[5];
    const float* b2          = (const float*)d_in[6];
    const float* W3          = (const float*)d_in[7];
    const float* b3          = (const float*)d_in[8];
    float* out = (float*)d_out;

    f16*   wH = (f16*)d_ws;
    float* wF = (float*)((char*)d_ws + WS_F32_OFF);

    din_prep_weights<<<1, 256, 0, stream>>>(W1, b1, W2, b2, W3, b3, wH, wF);
    din_attention_kernel<<<Bsz, 256, 0, stream>>>(queries, keys, keys_length, wH, wF, out);
}
